// Embedding_44641890074678
// MI455X (gfx1250) — matrix operand order repaired and checked by reading, not compiled, not run on hardware
//
#include <hip/hip_runtime.h>
#include <hip/hip_bf16.h>

// ---------------------------------------------------------------------------
// Fused (Linear + ReLU + LayerNorm) over three splits, bf16x3 WMMA GEMM.
// out[b, q, d], q in [0,97), d in [0,512).
// ---------------------------------------------------------------------------

typedef __attribute__((ext_vector_type(4)))  float   v4f;
typedef __attribute__((ext_vector_type(8)))  float   v8f;
typedef __attribute__((ext_vector_type(8)))  __bf16  v8bf;
typedef __attribute__((ext_vector_type(16))) __bf16  v16bf;

// ---------------------------------------------------------------------------
// Preprocess: W[L,512] f32  ->  bf16 hi/lo fragments in WMMA B-operand order.
// Fragment f = kt*32 + nt  (kt = K-tile of 32, nt = N-tile of 16).
// Within a fragment: lane n(0..15) = column nt*16+n, K = kt*32 + 0..15
//                    lane n(16..31) = column nt*16+(n-16), K = kt*32 + 16..31
// 16 consecutive bf16 per lane (contiguous K = 16h + s, the same order the A fragments use).
// ---------------------------------------------------------------------------
template <int SPLIT>
__global__ __launch_bounds__(256) void prep_w(const float* __restrict__ W,
                                              __bf16* __restrict__ whi,
                                              __bf16* __restrict__ wlo) {
  constexpr int L  = (SPLIT == 0) ? 400 : 256;
  constexpr int KT = (L + 31) / 32;
  const int gid = blockIdx.x * 256 + threadIdx.x;
  if (gid >= KT * 32 * 32) return;
  const int lane = gid & 31;
  const int fid  = gid >> 5;
  const int kt   = fid >> 5;
  const int nt   = fid & 31;
  const int col  = nt * 16 + (lane & 15);
  const int kb   = kt * 32 + ((lane & 16) ? 16 : 0);

  __bf16 hv[16], lv[16];
#pragma unroll
  for (int e = 0; e < 16; ++e) {
    const int k   = kb + e;
    const float v = (k < L) ? W[k * 512 + col] : 0.0f;
    const __bf16 h = (__bf16)v;
    hv[e] = h;
    lv[e] = (__bf16)(v - (float)h);
  }
#pragma unroll
  for (int e = 0; e < 16; ++e) {
    whi[(size_t)gid * 16 + e] = hv[e];
    wlo[(size_t)gid * 16 + e] = lv[e];
  }
}

// ---------------------------------------------------------------------------
// Fused GEMM + bias + ReLU + LayerNorm.
// Block: 256 threads (8 waves). Tile: 32 rows x 512 cols.
// Wave w owns columns [w*64, w*64+64) for all 32 rows -> acc[2][4] of v8f.
// bf16x3: acc += Ahi*Bhi + Ahi*Blo + Alo*Bhi  (f32 accumulate).
// ---------------------------------------------------------------------------
template <int SPLIT>
__global__ __launch_bounds__(256) void fused_fc_ln(
    const float* __restrict__ x, const float* __restrict__ bias,
    const float* __restrict__ gamma, const float* __restrict__ beta,
    const __bf16* __restrict__ whi, const __bf16* __restrict__ wlo,
    float* __restrict__ out) {
  constexpr int L     = (SPLIT == 0) ? 400 : 256;
  constexpr int KSL   = (SPLIT == 0) ? 1 : (SPLIT == 1) ? 60 : 36;   // slices
  constexpr int XOFF  = (SPLIT == 0) ? 0 : (SPLIT == 1) ? 400 : 15760;
  constexpr int QOFF  = (SPLIT == 0) ? 0 : (SPLIT == 1) ? 1 : 61;
  constexpr int KT    = (L + 31) / 32;
  constexpr int TOTAL = 24976;
  constexpr int D     = 512;
  constexpr int Q     = 97;
  constexpr int AP    = 40;  // padded LDS A row stride (bf16) -> 80B, no bank conflicts

  __shared__ __bf16 sAhi[32 * AP];
  __shared__ __bf16 sAlo[32 * AP];
  __shared__ float  sBias[D], sG[D], sBe[D];
  __shared__ float  sSum[32], sSsq[32];

  const int tid   = threadIdx.x;
  const int lane  = tid & 31;
  const int wave  = tid >> 5;
  const int lo16  = lane & 15;
  const int hiH   = (lane >> 4) & 1;  // which 16-lane half
  const int tileM = blockIdx.x * 32;

  // Stage bias / gamma / beta; zero LN accumulators.
  for (int i = tid; i < D; i += 256) {
    sBias[i] = bias[i];
    sG[i]    = gamma[i];
    sBe[i]   = beta[i];
  }
  if (tid < 32) { sSum[tid] = 0.0f; sSsq[tid] = 0.0f; }

  // A-tile cooperative load coords: thread -> (row 0..31, 4 consecutive K).
  const int ar = tid >> 3;
  const int ac = (tid & 7) << 2;
  const int am = tileM + ar;
  const int ab = am / KSL;            // batch index
  const int ak = am - ab * KSL;       // slice index within split
  const float* __restrict__ arow = x + (size_t)ab * TOTAL + XOFF + ak * L;

  const int akb = hiH ? 16 : 0;       // lane half K-chunk base (contiguous K = 16h + s)
  const int wcb = wave * 64;          // wave column base

  v8f acc[2][4];
#pragma unroll
  for (int mt = 0; mt < 2; ++mt)
#pragma unroll
    for (int j = 0; j < 4; ++j) acc[mt][j] = v8f{};

  for (int kt = 0; kt < KT; ++kt) {
    const int kk = kt * 32;
    __syncthreads();

    // ---- stage A tile (f32 -> bf16 hi/lo) into LDS ----
    float av[4];
    if constexpr ((L % 32) == 0) {
      const v4f t = *(const v4f*)(arow + kk + ac);
      av[0] = t[0]; av[1] = t[1]; av[2] = t[2]; av[3] = t[3];
    } else {
#pragma unroll
      for (int e = 0; e < 4; ++e) {
        const int k = kk + ac + e;
        av[e] = (k < L) ? arow[k] : 0.0f;
      }
    }
#pragma unroll
    for (int e = 0; e < 4; ++e) {
      const __bf16 h = (__bf16)av[e];
      sAhi[ar * AP + ac + e] = h;
      sAlo[ar * AP + ac + e] = (__bf16)(av[e] - (float)h);
    }
    __syncthreads();

    // Prefetch next K-step's B fragments (global_prefetch_b8, L2-resident W).
    if (kt + 1 < KT) {
      const size_t nidx = ((size_t)((kt + 1) * 32 + (wave << 2)) * 32 + lane) * 16;
      __builtin_prefetch(whi + nidx, 0, 1);
      __builtin_prefetch(wlo + nidx, 0, 1);
    }

    // ---- assemble A fragments from LDS (contiguous K = 16h + s, same order as B) ----
    v16bf ah[2], al[2];
#pragma unroll
    for (int mt = 0; mt < 2; ++mt) {
      const int r = mt * 16 + lo16;
      const v8bf h0 = *(const v8bf*)&sAhi[r * AP + akb];
      const v8bf h1 = *(const v8bf*)&sAhi[r * AP + akb + 8];
      const v8bf l0 = *(const v8bf*)&sAlo[r * AP + akb];
      const v8bf l1 = *(const v8bf*)&sAlo[r * AP + akb + 8];
      ah[mt] = __builtin_shufflevector(h0, h1, 0, 1, 2, 3, 4, 5, 6, 7, 8, 9, 10, 11, 12, 13, 14, 15);
      al[mt] = __builtin_shufflevector(l0, l1, 0, 1, 2, 3, 4, 5, 6, 7, 8, 9, 10, 11, 12, 13, 14, 15);
    }

    // ---- B fragments (pre-tiled, coalesced 32B/lane) + 24 WMMAs ----
#pragma unroll
    for (int j = 0; j < 4; ++j) {
      const size_t idx = ((size_t)(kt * 32 + (wave << 2) + j) * 32 + lane) * 16;
      const v16bf bh = *(const v16bf*)(whi + idx);
      const v16bf bl = *(const v16bf*)(wlo + idx);
#pragma unroll
      for (int mt = 0; mt < 2; ++mt) {
        acc[mt][j] = __builtin_amdgcn_wmma_f32_16x16x32_bf16(
            false, ah[mt], false, bh, (short)0, acc[mt][j], false, false);
        acc[mt][j] = __builtin_amdgcn_wmma_f32_16x16x32_bf16(
            false, ah[mt], false, bl, (short)0, acc[mt][j], false, false);
        acc[mt][j] = __builtin_amdgcn_wmma_f32_16x16x32_bf16(
            false, al[mt], false, bh, (short)0, acc[mt][j], false, false);
      }
    }
  }

  // ---- epilogue: bias + ReLU, per-row partial sums ----
  float s1[2][8], s2[2][8];
#pragma unroll
  for (int mt = 0; mt < 2; ++mt)
#pragma unroll
    for (int v = 0; v < 8; ++v) { s1[mt][v] = 0.0f; s2[mt][v] = 0.0f; }

#pragma unroll
  for (int mt = 0; mt < 2; ++mt)
#pragma unroll
    for (int j = 0; j < 4; ++j) {
      const int col  = wcb + (j << 4) + lo16;
      const float bb = sBias[col];
#pragma unroll
      for (int v = 0; v < 8; ++v) {
        float hv = acc[mt][j][v] + bb;
        hv = fmaxf(hv, 0.0f);
        acc[mt][j][v] = hv;
        s1[mt][v] += hv;
        s2[mt][v] += hv * hv;
      }
    }

  // Cross-lane reduction within each 16-lane half (same row set per half).
#pragma unroll
  for (int mt = 0; mt < 2; ++mt)
#pragma unroll
    for (int v = 0; v < 8; ++v) {
#pragma unroll
      for (int off = 8; off >= 1; off >>= 1) {
        s1[mt][v] += __shfl_xor(s1[mt][v], off, 16);
        s2[mt][v] += __shfl_xor(s2[mt][v], off, 16);
      }
    }
  if (lo16 == 0) {
    const int rb = hiH ? 8 : 0;
#pragma unroll
    for (int mt = 0; mt < 2; ++mt)
#pragma unroll
      for (int v = 0; v < 8; ++v) {
        atomicAdd(&sSum[mt * 16 + rb + v], s1[mt][v]);   // ds_add_f32
        atomicAdd(&sSsq[mt * 16 + rb + v], s2[mt][v]);
      }
  }
  __syncthreads();

  // ---- normalize, scale/shift, store ----
#pragma unroll
  for (int mt = 0; mt < 2; ++mt) {
#pragma unroll
    for (int v = 0; v < 8; ++v) {
      const int rl    = mt * 16 + (hiH ? 8 : 0) + v;
      const float mu  = sSum[rl] * (1.0f / 512.0f);
      const float var = sSsq[rl] * (1.0f / 512.0f) - mu * mu;
      const float inv = rsqrtf(var + 1e-5f);
      const int m  = tileM + rl;
      const int bI = m / KSL;
      const int kq = m - bI * KSL;
      float* __restrict__ orow = out + ((size_t)bI * Q + QOFF + kq) * D;
#pragma unroll
      for (int j = 0; j < 4; ++j) {
        const int col = wcb + (j << 4) + lo16;
        orow[col] = (acc[mt][j][v] - mu) * inv * sG[col] + sBe[col];
      }
    }
  }
}

// ---------------------------------------------------------------------------
// Workspace layout (bytes):
//   whi0 @ 0        (13*32*32*16*2 = 425984)
//   whi1 @ 425984   (8*32*32*16*2  = 262144)
//   whi2 @ 688128   (262144)
//   wlo0 @ 950272   (425984)
//   wlo1 @ 1376256  (262144)
//   wlo2 @ 1638400  (262144)   total = 1900544 B
// ---------------------------------------------------------------------------
extern "C" void kernel_launch(void* const* d_in, const int* in_sizes, int n_in,
                              void* d_out, int out_size, void* d_ws, size_t ws_size,
                              hipStream_t stream) {
  (void)in_sizes; (void)n_in; (void)out_size; (void)ws_size;

  const float* x   = (const float*)d_in[0];
  const float* W0  = (const float*)d_in[1];
  const float* b0  = (const float*)d_in[2];
  const float* g0  = (const float*)d_in[3];
  const float* be0 = (const float*)d_in[4];
  const float* W1  = (const float*)d_in[5];
  const float* b1  = (const float*)d_in[6];
  const float* g1  = (const float*)d_in[7];
  const float* be1 = (const float*)d_in[8];
  const float* W2  = (const float*)d_in[9];
  const float* b2  = (const float*)d_in[10];
  const float* g2  = (const float*)d_in[11];
  const float* be2 = (const float*)d_in[12];
  float* out = (float*)d_out;

  char* ws = (char*)d_ws;
  __bf16* whi0 = (__bf16*)(ws + 0);
  __bf16* whi1 = (__bf16*)(ws + 425984);
  __bf16* whi2 = (__bf16*)(ws + 688128);
  __bf16* wlo0 = (__bf16*)(ws + 950272);
  __bf16* wlo1 = (__bf16*)(ws + 1376256);
  __bf16* wlo2 = (__bf16*)(ws + 1638400);

  // Weight preprocessing (tiny; re-run every call, deterministic).
  prep_w<0><<<52, 256, 0, stream>>>(W0, whi0, wlo0);  // 13*32*32 = 13312 threads
  prep_w<1><<<32, 256, 0, stream>>>(W1, whi1, wlo1);  // 8192 threads
  prep_w<2><<<32, 256, 0, stream>>>(W2, whi2, wlo2);

  // Fused GEMM+ReLU+LN: one block per 32 flattened rows.
  fused_fc_ln<0><<<64, 256, 0, stream>>>(x, b0, g0, be0, whi0, wlo0, out);    // M=2048
  fused_fc_ln<1><<<3840, 256, 0, stream>>>(x, b1, g1, be1, whi1, wlo1, out);  // M=122880
  fused_fc_ln<2><<<2304, 256, 0, stream>>>(x, b2, g2, be2, whi2, wlo2, out);  // M=73728
}
